// AttnBlock_75024488727145
// MI455X (gfx1250) — matrix operand order repaired and checked by reading, not compiled, not run on hardware
//
#include <hip/hip_runtime.h>
#include <hip/hip_bf16.h>

typedef _Float16 v16h __attribute__((ext_vector_type(16)));
typedef _Float16 v8h  __attribute__((ext_vector_type(8)));
typedef float    v8f  __attribute__((ext_vector_type(8)));

#define C_DIM 256
#define N_SP  4096

__device__ __forceinline__ v16h make_frag(v8h lo, v8h hi) {
  v16h r;
#pragma unroll
  for (int i = 0; i < 8; ++i) { r[i] = lo[i]; r[8 + i] = hi[i]; }
  return r;
}

// A fragment: 16x32 f16. Storage: rows of M with K contiguous (lda halfs/row).
// ISA layout: lane l (half-group h=l>>4) holds row M=l&15,
//   halfs 0..7  = K[k0 + h*8 .. +7], halfs 8..15 = K[k0 + 16 + h*8 .. +7]
__device__ __forceinline__ v16h load_frag_a(const _Float16* __restrict__ A,
                                            int lda, int m0, int k0) {
  const int l = threadIdx.x & 31;
  const _Float16* p = A + (size_t)(m0 + (l & 15)) * lda + k0 + ((l >> 4) << 3);
  v8h lo = *(const v8h*)p;
  v8h hi = *(const v8h*)(p + 16);
  return make_frag(lo, hi);
}

// B fragment: 32x16 f16 (KxN). Storage: (N, K) row-major, K contiguous.
// ISA layout: lane l holds column N=l&15, halfs 0..7 = K[k0 + h*8 .. +7], halfs 8..15 = K[k0 + 16 + h*8 .. +7].
__device__ __forceinline__ v16h load_frag_b(const _Float16* __restrict__ B,
                                            int ldb, int n0, int k0) {
  const int l = threadIdx.x & 31;
  const _Float16* p = B + (size_t)(n0 + (l & 15)) * ldb + k0 + ((l >> 4) << 3);
  v8h lo = *(const v8h*)p;
  v8h hi = *(const v8h*)(p + 16);
  return make_frag(lo, hi);
}

// Generic NT GEMM: D(M,N) = A(M,K) * B(N,K)^T. Wave tile 16(M) x 64(N).
// EPI 0: f16 store + optional bias (by N if bias_by_n else by M)
// EPI 1: f32 store * scale
// EPI 2: f32 store + bias[M] + residual X
template <int EPI>
__global__ void gemm_nt(const _Float16* __restrict__ A, long long sA, int lda,
                        const _Float16* __restrict__ B, long long sB, int ldb,
                        void* __restrict__ Dv, long long sD, int ldd,
                        const float* __restrict__ bias, int bias_by_n,
                        const float* __restrict__ X, long long sX,
                        float scale, int M, int N, int K) {
  const int wave = threadIdx.x >> 5;
  const int tiles_n = N >> 6;
  const int tile = blockIdx.x * 8 + wave;
  if (tile >= (M >> 4) * tiles_n) return;          // wave-uniform: EXEC stays full
  const int m0 = (tile / tiles_n) << 4;
  const int n0 = (tile % tiles_n) << 6;
  const int b = blockIdx.y;
  const _Float16* Ab = A + (size_t)b * sA;
  const _Float16* Bb = B + (size_t)b * sB;

  v8f acc[4] = {};
  for (int k0 = 0; k0 < K; k0 += 32) {
    v16h a = load_frag_a(Ab, lda, m0, k0);
#pragma unroll
    for (int j = 0; j < 4; ++j) {
      v16h bf = load_frag_b(Bb, ldb, n0 + (j << 4), k0);
      acc[j] = __builtin_amdgcn_wmma_f32_16x16x32_f16(
          false, a, false, bf, (short)0, acc[j], false, false);
    }
  }

  const int l  = threadIdx.x & 31;
  const int nn = l & 15;
  const int mh = (l >> 4) << 3;
#pragma unroll
  for (int j = 0; j < 4; ++j) {
#pragma unroll
    for (int r = 0; r < 8; ++r) {
      const int m = m0 + mh + r;
      const int n = n0 + (j << 4) + nn;
      float v = acc[j][r];
      if (EPI == 0) {
        if (bias) v += bias_by_n ? bias[n] : bias[m];
        (((_Float16*)Dv) + (size_t)b * sD)[(size_t)m * ldd + n] = (_Float16)v;
      } else if (EPI == 1) {
        (((float*)Dv) + (size_t)b * sD)[(size_t)m * ldd + n] = v * scale;
      } else {
        float* D = ((float*)Dv) + (size_t)b * sD;
        v += bias[m] + X[(size_t)b * sX + (size_t)m * ldd + n];
        D[(size_t)m * ldd + n] = v;
      }
    }
  }
}

// hv_t(n,c) = attn(n,m) * V(c,m)^T, attn read transposed from fp32 (m,n) buffer
__global__ void gemm_attn_v(const float* __restrict__ Att, long long sAtt, int ldn,
                            const _Float16* __restrict__ Vcm, long long sV, int ldb,
                            _Float16* __restrict__ Hvt, long long sH, int ldd,
                            int M, int N, int K) {
  const int wave = threadIdx.x >> 5;
  const int tiles_n = N >> 6;
  const int tile = blockIdx.x * 8 + wave;
  if (tile >= (M >> 4) * tiles_n) return;
  const int m0 = (tile / tiles_n) << 4;
  const int n0 = (tile % tiles_n) << 6;
  const int b = blockIdx.y;
  const float* att = Att + (size_t)b * sAtt;
  const _Float16* V = Vcm + (size_t)b * sV;

  const int l   = threadIdx.x & 31;
  const int col = m0 + (l & 15);        // row of attn(n,.) == column of (m,n) storage
  const int kh  = (l >> 4) << 3;

  v8f acc[4] = {};
  for (int k0 = 0; k0 < K; k0 += 32) {
    v16h a;
#pragma unroll
    for (int j2 = 0; j2 < 8; ++j2) {
      a[j2]     = (_Float16)att[(size_t)(k0 + kh + j2) * ldn + col];
      a[8 + j2] = (_Float16)att[(size_t)(k0 + 16 + kh + j2) * ldn + col];
    }
#pragma unroll
    for (int j = 0; j < 4; ++j) {
      v16h bf = load_frag_b(V, ldb, n0 + (j << 4), k0);
      acc[j] = __builtin_amdgcn_wmma_f32_16x16x32_f16(
          false, a, false, bf, (short)0, acc[j], false, false);
    }
  }
  const int nn = l & 15;
  const int mh = (l >> 4) << 3;
  _Float16* D = Hvt + (size_t)b * sH;
#pragma unroll
  for (int j = 0; j < 4; ++j)
#pragma unroll
    for (int r = 0; r < 8; ++r)
      D[(size_t)(m0 + mh + r) * ldd + n0 + (j << 4) + nn] = (_Float16)acc[j][r];
}

__global__ void groupnorm_kernel(const float* __restrict__ x,
                                 const float* __restrict__ gamma,
                                 const float* __restrict__ beta,
                                 _Float16* __restrict__ hn_t) {
  const int g = blockIdx.x;   // 32 groups of 8 channels
  const int b = blockIdx.y;
  const int tid = threadIdx.x;
  const float* xg = x + ((size_t)b * C_DIM + g * 8) * N_SP;

  float s = 0.f, ss = 0.f;
  for (int i = tid; i < 8 * N_SP; i += 256) {
    float v = xg[i];
    s += v; ss += v * v;
  }
  __shared__ float sh_s[256], sh_q[256];
  sh_s[tid] = s; sh_q[tid] = ss;
  __syncthreads();
  for (int off = 128; off > 0; off >>= 1) {
    if (tid < off) { sh_s[tid] += sh_s[tid + off]; sh_q[tid] += sh_q[tid + off]; }
    __syncthreads();
  }
  const float inv_n = 1.0f / (8.0f * N_SP);
  const float mu  = sh_s[0] * inv_n;
  const float var = sh_q[0] * inv_n - mu * mu;
  const float rs  = rsqrtf(var + 1e-6f);

  _Float16* out = hn_t + (size_t)b * N_SP * C_DIM;
  for (int i = tid; i < 8 * N_SP; i += 256) {
    const int cl = i >> 12;             // local channel
    const int n  = i & (N_SP - 1);
    const int c  = g * 8 + cl;
    const float v = (xg[i] - mu) * rs * gamma[c] + beta[c];
    out[(size_t)n * C_DIM + c] = (_Float16)v;
  }
}

__global__ void f32_to_f16(const float* __restrict__ src,
                           _Float16* __restrict__ dst, int n) {
  const int i = blockIdx.x * blockDim.x + threadIdx.x;
  if (i < n) dst[i] = (_Float16)src[i];
}

// In-place column softmax over m on attn^T stored (m,n); coalesced across n.
__global__ void softmax_col(float* __restrict__ att) {
  const int n = blockIdx.x * 256 + threadIdx.x;
  float* p = att + (size_t)blockIdx.y * N_SP * N_SP + n;
  float mx = -3.402823466e38f, sum = 0.f;
  for (int m = 0; m < N_SP; ++m) {
    const float v = p[(size_t)m * N_SP];
    const float nm = fmaxf(mx, v);
    sum = sum * __expf(mx - nm) + __expf(v - nm);
    mx = nm;
  }
  const float inv = 1.0f / sum;
  for (int m = 0; m < N_SP; ++m) {
    const float v = p[(size_t)m * N_SP];
    p[(size_t)m * N_SP] = __expf(v - mx) * inv;
  }
}

extern "C" void kernel_launch(void* const* d_in, const int* in_sizes, int n_in,
                              void* d_out, int out_size, void* d_ws, size_t ws_size,
                              hipStream_t stream) {
  const float* x  = (const float*)d_in[0];
  const float* gg = (const float*)d_in[1];
  const float* gb = (const float*)d_in[2];
  const float* wq = (const float*)d_in[3];
  const float* bq = (const float*)d_in[4];
  const float* wk = (const float*)d_in[5];
  const float* bk = (const float*)d_in[6];
  const float* wv = (const float*)d_in[7];
  const float* bv = (const float*)d_in[8];
  const float* wp = (const float*)d_in[9];
  const float* bp = (const float*)d_in[10];

  // workspace layout (all f16): 4 weights + 5 staging matrices, ~20.5 MB total
  _Float16* ws = (_Float16*)d_ws;
  const size_t WSZ = (size_t)C_DIM * C_DIM;     // 65536
  const size_t MSZ = (size_t)2 * N_SP * C_DIM;  // 2,097,152 halfs per matrix
  _Float16* wq16 = ws;
  _Float16* wk16 = wq16 + WSZ;
  _Float16* wv16 = wk16 + WSZ;
  _Float16* wp16 = wv16 + WSZ;
  _Float16* hn_t = wp16 + WSZ;    // (b, n, c)
  _Float16* q_t  = hn_t + MSZ;    // (b, n, c)
  _Float16* k_t  = q_t  + MSZ;    // (b, m, c)
  _Float16* v_cm = k_t  + MSZ;    // (b, c, m)
  _Float16* hv_t = v_cm + MSZ;    // (b, n, c)

  float* out = (float*)d_out;                       // (b, c, n) residual output
  float* att = out + (size_t)2 * C_DIM * N_SP;      // (b, m, n) == attn^T, fp32

  const long long sHN  = (long long)N_SP * C_DIM;   // 1,048,576
  const long long sATT = (long long)N_SP * N_SP;    // 16,777,216

  f32_to_f16<<<256, 256, 0, stream>>>(wq, wq16, (int)WSZ);
  f32_to_f16<<<256, 256, 0, stream>>>(wk, wk16, (int)WSZ);
  f32_to_f16<<<256, 256, 0, stream>>>(wv, wv16, (int)WSZ);
  f32_to_f16<<<256, 256, 0, stream>>>(wp, wp16, (int)WSZ);

  groupnorm_kernel<<<dim3(32, 2), 256, 0, stream>>>(x, gg, gb, hn_t);

  // q_t(n,o) = hn_t(n,c) * wq(o,c)^T + bq[o]
  gemm_nt<0><<<dim3(128, 2), 256, 0, stream>>>(hn_t, sHN, C_DIM, wq16, 0, C_DIM,
      q_t, sHN, C_DIM, bq, 1, nullptr, 0, 1.f, N_SP, C_DIM, C_DIM);
  gemm_nt<0><<<dim3(128, 2), 256, 0, stream>>>(hn_t, sHN, C_DIM, wk16, 0, C_DIM,
      k_t, sHN, C_DIM, bk, 1, nullptr, 0, 1.f, N_SP, C_DIM, C_DIM);
  // v_cm(o,m) = wv(o,c) * hn_t(m,c)^T + bv[o]
  gemm_nt<0><<<dim3(128, 2), 256, 0, stream>>>(wv16, 0, C_DIM, hn_t, sHN, C_DIM,
      v_cm, sHN, N_SP, bv, 0, nullptr, 0, 1.f, C_DIM, N_SP, C_DIM);
  // attn^T logits (m,n) = k_t(m,c) * q_t(n,c)^T * c^-1/2  -> directly into d_out
  gemm_nt<1><<<dim3(2048, 2), 256, 0, stream>>>(k_t, sHN, C_DIM, q_t, sHN, C_DIM,
      att, sATT, N_SP, nullptr, 0, nullptr, 0, 0.0625f, N_SP, N_SP, C_DIM);

  softmax_col<<<dim3(16, 2), 256, 0, stream>>>(att);

  // hv_t(n,c) = attn(n,m) * v_cm(c,m)^T
  gemm_attn_v<<<dim3(128, 2), 256, 0, stream>>>(att, sATT, N_SP, v_cm, sHN, N_SP,
      hv_t, sHN, C_DIM, N_SP, C_DIM, N_SP);
  // out(o,n) = wp(o,c) * hv_t(n,c)^T + bp[o] + x
  gemm_nt<2><<<dim3(128, 2), 256, 0, stream>>>(wp16, 0, C_DIM, hv_t, sHN, C_DIM,
      out, sHN, N_SP, bp, 0, x, sHN, 1.f, C_DIM, N_SP, C_DIM);
}
